// GraphAttentionLayer_41936060678285
// MI455X (gfx1250) — matrix operand order repaired and checked by reading, not compiled, not run on hardware
//
#include <hip/hip_runtime.h>
#include <hip/hip_bf16.h>

typedef __attribute__((ext_vector_type(16))) _Float16 v16h;
typedef __attribute__((ext_vector_type(8)))  _Float16 half8;
typedef __attribute__((ext_vector_type(8)))  float    v8f;

#define BATCH  8
#define NNODE  2048
#define FIN    128
#define FOUT   64
#define ALPHA  0.2f
#define LOG2E  1.4426950408889634f
#define NWAVE  8                       // j-split waves per i-tile
#define JCHUNK (NNODE / NWAVE)         // 256 j's per wave
#define PFD    128                     // adj prefetch distance (ints = 4 iters)

// order-preserving float<->uint encoding for atomicMax on floats
__device__ inline unsigned enc_f(float f) {
    unsigned u = __float_as_uint(f);
    return (u & 0x80000000u) ? ~u : (u | 0x80000000u);
}
__device__ inline float dec_f(unsigned u) {
    return __uint_as_float((u & 0x80000000u) ? (u & 0x7FFFFFFFu) : ~u);
}

__global__ void gat_init_kernel(unsigned* __restrict__ sjmax_enc)
{
    if (threadIdx.x < BATCH) sjmax_enc[threadIdx.x] = 0u;   // < enc(-inf)
}

// ---------------------------------------------------------------------------
// Kernel 1: Wh = h @ W (f32), WhT (f16, [b][f][n] for WMMA-B contiguity),
//           si = Wh@a1, sj = Wh@a2, per-batch max_j sj.  One wave per row.
// ---------------------------------------------------------------------------
__global__ __launch_bounds__(256)
void gat_proj_kernel(const float* __restrict__ h,
                     const float* __restrict__ W,
                     const float* __restrict__ a,
                     _Float16* __restrict__ whT,
                     float* __restrict__ si,
                     float* __restrict__ sj,
                     unsigned* __restrict__ sjmax_enc)
{
    __shared__ float hrow[8][FIN];
    const int wv   = threadIdx.x >> 5;
    const int lane = threadIdx.x & 31;
    const int row  = blockIdx.x * 8 + wv;          // 0 .. B*N-1
    const int b    = row >> 11;                    // /2048
    const int n    = row & (NNODE - 1);

    const float* hp = h + (size_t)row * FIN;
    hrow[wv][lane      ] = hp[lane      ];
    hrow[wv][lane + 32 ] = hp[lane + 32 ];
    hrow[wv][lane + 64 ] = hp[lane + 64 ];
    hrow[wv][lane + 96 ] = hp[lane + 96 ];
    // same-wave LDS RAW: compiler inserts s_wait_dscnt

    const int f0 = lane, f1 = lane + 32;
    float wh0 = 0.f, wh1 = 0.f;
#pragma unroll 8
    for (int k = 0; k < FIN; ++k) {
        const float hk = hrow[wv][k];
        wh0 = fmaf(hk, W[k * FOUT + f0], wh0);
        wh1 = fmaf(hk, W[k * FOUT + f1], wh1);
    }

    whT[((size_t)(b * FOUT + f0)) * NNODE + n] = (_Float16)wh0;
    whT[((size_t)(b * FOUT + f1)) * NNODE + n] = (_Float16)wh1;

    float sip = wh0 * a[f0]        + wh1 * a[f1];
    float sjp = wh0 * a[FOUT + f0] + wh1 * a[FOUT + f1];
#pragma unroll
    for (int off = 16; off; off >>= 1) {
        sip += __shfl_xor(sip, off);
        sjp += __shfl_xor(sjp, off);
    }
    if (lane == 0) {
        si[row] = sip;
        sj[row] = sjp;
        atomicMax(&sjmax_enc[b], enc_f(sjp));
    }
}

// Load two 8-f16 runs (p..p+7, p+16..p+23) as a WMMA-B operand fragment
__device__ inline v16h load_b16(const _Float16* __restrict__ p)
{
    half8 lo = *(const half8*)p;
    half8 hi = *(const half8*)(p + 16);
    v16h r;
#pragma unroll
    for (int e = 0; e < 8; ++e) { r[e] = lo[e]; r[e + 8] = hi[e]; }
    return r;
}

// ---------------------------------------------------------------------------
// Kernel 2: fused masked-softmax attention + attn@Wh.
// Fixed per-row exp-shift eb = lrelu(si + max_j sj) >= all logits; loop-carried
// state is only the WMMA accumulators + a lane-local sum. 8 waves per i-tile,
// each streaming a 256-wide j-chunk with software prefetch; merged in LDS.
// ---------------------------------------------------------------------------
__global__ __launch_bounds__(32 * NWAVE)
void gat_attn_kernel(const int* __restrict__ adj,
                     const _Float16* __restrict__ whT,
                     const float* __restrict__ si,
                     const float* __restrict__ sj,
                     const unsigned* __restrict__ sjmax_enc,
                     float* __restrict__ out)
{
    __shared__ float accS[16][FOUT];   // merged numerator, 4KB
    __shared__ float sumS[16];         // merged denominator

    const int tid  = threadIdx.x;
    const int wv   = tid >> 5;
    const int lane = tid & 31;
    const int m    = lane & 15;        // row within i-tile (A-matrix M)
    const int hi   = lane >> 4;        // K-half selector per A layout
    const int tiles_per_b = NNODE / 16;
    const int b    = blockIdx.x / tiles_per_b;
    const int i0   = (blockIdx.x % tiles_per_b) * 16;

    for (int idx = tid; idx < 16 * FOUT; idx += 32 * NWAVE) ((float*)accS)[idx] = 0.f;
    if (tid < 16) sumS[tid] = 0.f;
    __syncthreads();

    const float si_m = si[b * NNODE + i0 + m];
    const float smax = dec_f(sjmax_enc[b]);
    float eb = si_m + smax;                        // upper bound on this row's logits
    eb = fmaxf(eb, ALPHA * eb);                    // lrelu (monotone)
    const float neb2 = -eb * LOG2E;                // folded exp shift (base-2)

    const int*      adj_row = adj + ((size_t)(b * NNODE + i0 + m)) * NNODE;
    const float*    sj_b    = sj + b * NNODE;
    const _Float16* whT_b   = whT + (size_t)b * FOUT * NNODE;

    v8f acc0 = {}, acc1 = {}, acc2 = {}, acc3 = {};
    float run_sum = 0.f;                           // lane-local half-row sum

    const int jbeg = wv * JCHUNK;
    const int jend = jbeg + JCHUNK;
    for (int j0 = jbeg; j0 < jend; j0 += 32) {
        const int kbase = j0 + hi * 8;
        const _Float16* bp = whT_b + (size_t)m * NNODE + j0 + hi * 8;

        // software prefetch: adj stream 4 iters ahead, WhT B-tiles 4 iters ahead
        __builtin_prefetch((const void*)(adj_row + kbase + PFD), 0, 1);
        __builtin_prefetch((const void*)(bp + PFD), 0, 3);
        __builtin_prefetch((const void*)(bp + (size_t)16 * NNODE + PFD), 0, 3);
        __builtin_prefetch((const void*)(bp + (size_t)32 * NNODE + PFD), 0, 3);
        __builtin_prefetch((const void*)(bp + (size_t)48 * NNODE + PFD), 0, 3);

        // adj + sj for this lane's 16 K-slots: two contiguous 32B runs each
        int4   am0 = *(const int4*)  (adj_row + kbase);
        int4   am1 = *(const int4*)  (adj_row + kbase + 4);
        int4   am2 = *(const int4*)  (adj_row + kbase + 16);
        int4   am3 = *(const int4*)  (adj_row + kbase + 20);
        float4 sv0 = *(const float4*)(sj_b    + kbase);
        float4 sv1 = *(const float4*)(sj_b    + kbase + 4);
        float4 sv2 = *(const float4*)(sj_b    + kbase + 16);
        float4 sv3 = *(const float4*)(sj_b    + kbase + 20);

        // B operands: column f = 16*t + m of WhT, rows j0+hi*8 .. +7 and j0+16+hi*8 .. +7
        v16h Bv0 = load_b16(bp);
        v16h Bv1 = load_b16(bp + (size_t)16 * NNODE);
        v16h Bv2 = load_b16(bp + (size_t)32 * NNODE);
        v16h Bv3 = load_b16(bp + (size_t)48 * NNODE);

        int   am[16] = { am0.x, am0.y, am0.z, am0.w, am1.x, am1.y, am1.z, am1.w,
                         am2.x, am2.y, am2.z, am2.w, am3.x, am3.y, am3.z, am3.w };
        float sv[16] = { sv0.x, sv0.y, sv0.z, sv0.w, sv1.x, sv1.y, sv1.z, sv1.w,
                         sv2.x, sv2.y, sv2.z, sv2.w, sv3.x, sv3.y, sv3.z, sv3.w };

        v16h A;
#pragma unroll
        for (int t = 0; t < 16; ++t) {
            float v = si_m + sv[t];
            v = fmaxf(v, ALPHA * v);                          // leaky relu
            float p = __builtin_amdgcn_exp2f(fmaf(v, LOG2E, neb2)); // exp(v - eb)
            p = (am[t] > 0) ? p : 0.f;                        // adjacency mask
            run_sum += p;
            A[t] = (_Float16)p;
        }

        acc0 = __builtin_amdgcn_wmma_f32_16x16x32_f16(false, A, false, Bv0, (short)0, acc0, false, false);
        acc1 = __builtin_amdgcn_wmma_f32_16x16x32_f16(false, A, false, Bv1, (short)0, acc1, false, false);
        acc2 = __builtin_amdgcn_wmma_f32_16x16x32_f16(false, A, false, Bv2, (short)0, acc2, false, false);
        acc3 = __builtin_amdgcn_wmma_f32_16x16x32_f16(false, A, false, Bv3, (short)0, acc3, false, false);
    }

    // ---- merge partial sums / accumulators (no corrections needed) ----
    atomicAdd(&sumS[m], run_sum);      // both K-halves contribute their part

#pragma unroll
    for (int r = 0; r < 8; ++r) {
        const int row = r + hi * 8;    // D layout: VGPR r, lanes>=16 are rows 8..15
        atomicAdd(&accS[row][m     ], acc0[r]);
        atomicAdd(&accS[row][m + 16], acc1[r]);
        atomicAdd(&accS[row][m + 32], acc2[r]);
        atomicAdd(&accS[row][m + 48], acc3[r]);
    }
    __syncthreads();

    // normalized cooperative store
    for (int idx = tid; idx < 16 * FOUT; idx += 32 * NWAVE) {
        const int row = idx >> 6;
        const int f   = idx & (FOUT - 1);
        const float denom = fmaxf(sumS[row], 1e-30f);
        out[((size_t)(b * NNODE + i0 + row)) * FOUT + f] = accS[row][f] / denom;
    }
}

// ---------------------------------------------------------------------------
extern "C" void kernel_launch(void* const* d_in, const int* in_sizes, int n_in,
                              void* d_out, int out_size, void* d_ws, size_t ws_size,
                              hipStream_t stream)
{
    const float* h   = (const float*)d_in[0];   // (8,2048,128) f32
    const int*   adj = (const int*)  d_in[1];   // (8,2048,2048) i32
    const float* W   = (const float*)d_in[2];   // (128,64) f32
    const float* a   = (const float*)d_in[3];   // (128,1) f32
    float* out = (float*)d_out;                 // (8,2048,64) f32

    _Float16* whT = (_Float16*)d_ws;                                   // 2 MB
    float* si = (float*)((char*)d_ws + (size_t)BATCH * FOUT * NNODE * sizeof(_Float16));
    float* sj = si + BATCH * NNODE;
    unsigned* sjmax_enc = (unsigned*)(sj + BATCH * NNODE);

    gat_init_kernel<<<1, 32, 0, stream>>>(sjmax_enc);
    gat_proj_kernel<<<(BATCH * NNODE) / 8, 256, 0, stream>>>(h, W, a, whT, si, sj, sjmax_enc);
    gat_attn_kernel<<<BATCH * (NNODE / 16), 32 * NWAVE, 0, stream>>>(adj, whT, si, sj, sjmax_enc, out);
}
